// Kernel_17772574670927
// MI455X (gfx1250) — matrix operand order repaired and checked by reading, not compiled, not run on hardware
//
#include <hip/hip_runtime.h>
#include <hip/hip_bf16.h>

typedef __attribute__((ext_vector_type(16))) __bf16 v16bf;
typedef __attribute__((ext_vector_type(8)))  float  v8f;

struct U8x32 { uint4 lo, hi; };              // 32 bytes
union FragCast { U8x32 u; v16bf v; };        // reinterpret 16 bf16 <-> 2x uint4

__device__ __forceinline__ unsigned pack_bf16x2(float a, float b) {
    // round-to-nearest-even f32 -> bf16, packed (a = low half, b = high half)
    union { float f; unsigned u; } ua, ub;
    ua.f = a; ub.f = b;
    unsigned x = (ua.u + 0x7fffu + ((ua.u >> 16) & 1u)) >> 16;
    unsigned y = (ub.u + 0x7fffu + ((ub.u >> 16) & 1u)) >> 16;
    return (y << 16) | (x & 0xffffu);
}

// ---------------------------------------------------------------------------
// Pack W[1:257] (self-h) and W[258:514] (nbr-h) rows into bf16 B-fragments.
// Fragment (kt, nt): 32 lanes x 8 dwords, dword v of lane l holds the bf16
// pair for K = kt*32 + ((l>=16)?16:0) + 2v, N = nt*16 + (l&15).
// Linear index: id = ((kt*16 + nt)*32 + l)*8 + v   (65536 dwords = 256 KB)
// ---------------------------------------------------------------------------
__global__ void pack_w_kernel(const float* __restrict__ W,
                              unsigned* __restrict__ wp) {
    int id = blockIdx.x * blockDim.x + threadIdx.x;   // 0 .. 65535
    int v  =  id        & 7;
    int l  = (id >> 3)  & 31;
    int nt = (id >> 8)  & 15;
    int kt =  id >> 12;
    int koff = ((l >> 4) << 4) + (v << 1);            // (l<16?0:16) + 2v (even)
    int k = kt * 32 + koff;
    int n = nt * 16 + (l & 15);
    int f0 = (k < 256) ? (1 + k) : (258 + (k - 256)); // W row for this K
    float lo = W[f0 * 256 + n];
    float hi = W[(f0 + 1) * 256 + n];
    wp[id] = pack_bf16x2(lo, hi);
}

// ---------------------------------------------------------------------------
// Main fused kernel: per block -> 32 rows x 256 cols of the output.
//   Stage 1: build bf16 A-tile X[32][512] in LDS:
//            X[:,0:256)  = 8 * node_h, X[:,256:512) = sum_n nbr_h
//   Stage 2: 8 waves, wave w: row tile r=w>>2, col group cg=w&3 (4 n-tiles)
//            16 K-steps of v_wmma_f32_16x16x32_bf16, f32 accumulate.
//   Epilogue: + 8*node_v*W[0,:] + (sum nbr_v)*W[257,:] + 8*bias, store.
// ---------------------------------------------------------------------------
#define LDS_PITCH_DW 260   // 512 bf16 + 8 pad  -> 1040 bytes (4-bank row skew)

__global__ __launch_bounds__(256) void msg_kernel(
    const float* __restrict__ node_v, const float* __restrict__ node_h,
    const float* __restrict__ nbr_v,  const float* __restrict__ nbr_h,
    const float* __restrict__ W,      const float* __restrict__ bias,
    const unsigned* __restrict__ wp,  float* __restrict__ out)
{
    __shared__ unsigned ldsX[32 * LDS_PITCH_DW];   // 33280 bytes
    __shared__ float rs_a[32];                     // 8 * node_v per row
    __shared__ float rs_b[32];                     // sum_n nbr_v per row

    const int t = threadIdx.x;
    const long long b0 = (long long)blockIdx.x * 32;

    // ---- Stage 1a: self part (x8), 32 rows x 128 bf16-pairs ----
    #pragma unroll 4
    for (int i = 0; i < 16; ++i) {
        int idx = t + i * 256;            // 0..4095
        int row = idx >> 7;
        int hp  = idx & 127;
        const float2 v = *(const float2*)(node_h + ((b0 + row) * 256 + hp * 2));
        ldsX[row * LDS_PITCH_DW + hp] = pack_bf16x2(8.0f * v.x, 8.0f * v.y);
    }
    // ---- Stage 1b: neighbor sum part ----
    #pragma unroll 2
    for (int i = 0; i < 16; ++i) {
        int idx = t + i * 256;
        int row = idx >> 7;
        int hp  = idx & 127;
        const float* base = nbr_h + ((b0 + row) * 8) * 256 + hp * 2;
        float sx = 0.f, sy = 0.f;
        #pragma unroll
        for (int n = 0; n < 8; ++n) {
            float2 v = *(const float2*)(base + n * 256);
            sx += v.x; sy += v.y;
        }
        ldsX[row * LDS_PITCH_DW + 128 + hp] = pack_bf16x2(sx, sy);
    }
    // ---- Stage 1c: per-row rank-1 scalars ----
    if (t < 32) {
        rs_a[t] = 8.0f * node_v[b0 + t];
        float s = 0.f;
        #pragma unroll
        for (int n = 0; n < 8; ++n) s += nbr_v[(b0 + t) * 8 + n];
        rs_b[t] = s;
    }
    __syncthreads();

    // ---- Stage 2: WMMA ----
    const int w  = t >> 5;          // wave 0..7
    const int l  = t & 31;          // lane
    const int r  = w >> 2;          // row tile (0..1) -> rows r*16 .. r*16+15
    const int cg = w & 3;           // col group -> cols cg*64 .. cg*64+63

    v8f acc[4] = {v8f{0.f}, v8f{0.f}, v8f{0.f}, v8f{0.f}};

    // A-fragment LDS base (bytes): row = r*16 + (l&15); hi-half lanes take K+16
    const int arow = r * 16 + (l & 15);
    const unsigned abase = (unsigned)arow * (LDS_PITCH_DW * 4) + ((l >> 4) ? 32u : 0u);
    const char* lds_bytes = (const char*)ldsX;

    for (int kt = 0; kt < 16; ++kt) {
        const char* ap = lds_bytes + abase + kt * 64;
        FragCast af;
        af.u.lo = *(const uint4*)(ap);        // K +0..7   (or +16..23 for hi lanes)
        af.u.hi = *(const uint4*)(ap + 16);   // K +8..15  (or +24..31)
        const v16bf afrag = af.v;
        #pragma unroll
        for (int j = 0; j < 4; ++j) {
            const int nt = cg * 4 + j;
            const uint4* bp = (const uint4*)(wp + (((kt * 16 + nt) * 32 + l) << 3));
            FragCast bf;
            bf.u.lo = bp[0];
            bf.u.hi = bp[1];
            acc[j] = __builtin_amdgcn_wmma_f32_16x16x32_bf16(
                /*neg_a=*/false, afrag, /*neg_b=*/false, bf.v,
                /*c_mod=*/(short)0, acc[j], /*reuse_a=*/false, /*reuse_b=*/false);
        }
    }

    // ---- Epilogue: rank-1 v-terms + bias, then store ----
    const int mhi = (l >> 4) << 3;  // lanes 16-31 hold rows M+8
    const int nlo = l & 15;
    #pragma unroll
    for (int j = 0; j < 4; ++j) {
        const int n   = cg * 64 + j * 16 + nlo;
        const float w0   = W[n];                // W row 0   (self v)
        const float w257 = W[257 * 256 + n];    // W row 257 (nbr v)
        const float bb   = 8.0f * bias[n];
        #pragma unroll
        for (int i = 0; i < 8; ++i) {
            const int mrow = r * 16 + i + mhi;  // row within block
            const float val = acc[j][i] + rs_a[mrow] * w0 + rs_b[mrow] * w257 + bb;
            out[(b0 + mrow) * 256 + n] = val;
        }
    }
}

// ---------------------------------------------------------------------------
extern "C" void kernel_launch(void* const* d_in, const int* in_sizes, int n_in,
                              void* d_out, int out_size, void* d_ws, size_t ws_size,
                              hipStream_t stream) {
    const float* node_v = (const float*)d_in[0];  // [B,1]
    const float* node_h = (const float*)d_in[1];  // [B,256]
    const float* nbr_v  = (const float*)d_in[2];  // [B,8,1]
    const float* nbr_h  = (const float*)d_in[3];  // [B,8,256]
    const float* W      = (const float*)d_in[4];  // [514,256]
    const float* bias   = (const float*)d_in[5];  // [256]
    float* out = (float*)d_out;                   // [B,256]
    unsigned* wp = (unsigned*)d_ws;               // 256 KB packed bf16 weights

    pack_w_kernel<<<256, 256, 0, stream>>>(W, wp);
    msg_kernel<<<65536 / 32, 256, 0, stream>>>(node_v, node_h, nbr_v, nbr_h,
                                               W, bias, wp, out);
}
